// RNN_64269890617992
// MI455X (gfx1250) — compile-verified, matrix operand order checked by reading, not run on hardware
//
#include <hip/hip_runtime.h>
#include <hip/hip_bf16.h>

// ---------------------------------------------------------------------------
// 2-layer LSTM for MI455X (gfx1250), wave32 + WMMA bf16 (fp32 accumulate).
// T=512, B=64, D=H=1024, NG=4096. Gate order (i,f,g,o) per PyTorch.
// ---------------------------------------------------------------------------

#define T_ 512
#define B_ 64
#define D_ 1024
#define H_ 1024
#define NG_ 4096

typedef __attribute__((ext_vector_type(16))) __bf16 v16bf;
typedef __attribute__((ext_vector_type(8)))  __bf16 v8bf;
typedef __attribute__((ext_vector_type(4)))  __bf16 v4bf;
typedef __attribute__((ext_vector_type(8)))  float  v8f;

// Load one 16x32 bf16 operand tile (A row-per-lane layout; B uses the
// symmetric column-per-lane layout == rows of the row-major [N,K] weight).
// Lane L (row = L&15) supplies K = k0 + kb .. k0+kb+7 in elements 0..7 and
// K = k0 + 16 + kb .. in elements 8..15, where kb = 8*(L>>4).
__device__ __forceinline__ v16bf ldtile(const __bf16* __restrict__ tbase,
                                        int ld, int k0, int lane) {
  const __bf16* p = tbase + (size_t)(lane & 15) * ld + k0 + ((lane >> 4) << 3);
  v8bf lo = *(const v8bf*)p;
  v8bf hi = *(const v8bf*)(p + 16);
  return __builtin_shufflevector(lo, hi, 0, 1, 2, 3, 4, 5, 6, 7,
                                 8, 9, 10, 11, 12, 13, 14, 15);
}

__device__ __forceinline__ float sigmoidf_(float x) {
  return 1.0f / (1.0f + __expf(-x));
}

// --------------------------- fp32 -> bf16 convert ---------------------------
__global__ void cvt_f32_bf16(const float* __restrict__ s, __bf16* __restrict__ d,
                             long n4) {
  long i = ((long)blockIdx.x * blockDim.x + threadIdx.x);
  if (i < n4) {
    const float4 v = ((const float4*)s)[i];
    v4bf o;
    o[0] = (__bf16)v.x; o[1] = (__bf16)v.y; o[2] = (__bf16)v.z; o[3] = (__bf16)v.w;
    ((v4bf*)d)[i] = o;
  }
}

// ------------------------------ state init ---------------------------------
__global__ void init_state(__bf16* __restrict__ h, float* __restrict__ c) {
  int i = blockIdx.x * blockDim.x + threadIdx.x;   // B_*H_ = 65536 threads
  h[i] = (__bf16)0.0f;
  c[i] = 0.0f;
}

// ---------------- input-projection GEMM: xg = A*W^T + b0 + b1 --------------
// A: [M=32768, K=1024] bf16 row-major; W: [N=4096, K=1024] bf16 row-major.
// One wave -> 64x64 fp32 tile (16 accumulators). Block = 8 waves (8 M-tiles).
__global__ __launch_bounds__(256) void gemm_xg(
    const __bf16* __restrict__ A, const __bf16* __restrict__ W,
    const float* __restrict__ b0, const float* __restrict__ b1,
    float* __restrict__ out) {
  const int lane = threadIdx.x & 31;
  const int wave = threadIdx.x >> 5;
  const int n0 = blockIdx.x * 64;
  const int m0 = (blockIdx.y * 8 + wave) * 64;

  v8f acc[4][4] = {};
  for (int k0 = 0; k0 < D_; k0 += 32) {
    v16bf a[4], b[4];
#pragma unroll
    for (int i = 0; i < 4; ++i)
      a[i] = ldtile(A + (size_t)(m0 + i * 16) * D_, D_, k0, lane);
#pragma unroll
    for (int i = 0; i < 4; ++i)
      b[i] = ldtile(W + (size_t)(n0 + i * 16) * D_, D_, k0, lane);
#pragma unroll
    for (int mi = 0; mi < 4; ++mi)
#pragma unroll
      for (int ni = 0; ni < 4; ++ni)
        acc[mi][ni] = __builtin_amdgcn_wmma_f32_16x16x32_bf16(
            false, a[mi], false, b[ni], (short)0, acc[mi][ni], false, false);
  }

  const int col_l = lane & 15;
  const int mh = (lane >> 4) * 8;
#pragma unroll
  for (int ni = 0; ni < 4; ++ni) {
    const int n = n0 + ni * 16 + col_l;
    const float bias = b0[n] + b1[n];
#pragma unroll
    for (int mi = 0; mi < 4; ++mi) {
      const int mrow = m0 + mi * 16 + mh;
#pragma unroll
      for (int r = 0; r < 8; ++r)
        out[(size_t)(mrow + r) * NG_ + n] = acc[mi][ni][r] + bias;
    }
  }
}

// ------------------------- fused recurrent step ----------------------------
// Grid: 64 WGs; WG wg owns h-columns [16*wg, 16*wg+16) of all 4 gates.
// Block: 512 threads = 16 waves = 4 gates x 4 K-quarters.
// LDS gate accumulator seeded with xg[t]; waves ds_add_f32 partial matmuls.
__global__ __launch_bounds__(512) void lstm_step(
    const float* __restrict__ xg_t,       // [B_, NG_]
    const __bf16* __restrict__ w_hh,      // [NG_, H_] bf16 row-major
    const __bf16* __restrict__ h_in,      // [B_, H_] bf16  (prev h, read-only)
    __bf16* __restrict__ h_out,           // [B_, H_] bf16  (next h)
    float* __restrict__ c_state,          // [B_, H_] fp32  (in/out, per-WG cols)
    __bf16* __restrict__ hseq_out,        // layer0: h1 sequence slot, else null
    float* __restrict__ hout_f32) {       // layer1: d_out slot, else null
  __shared__ float g_lds[4 * B_ * 16];    // 16 KB
  const int tid = threadIdx.x;
  const int wg = blockIdx.x;              // 0..63

  // 1) seed LDS with xg[t] for this WG's 4x(64x16) gate tiles (8 floats/thr)
  {
    const int e = tid * 8;
    const int gb = e >> 4;                // gate*64 + b
    const int half = (e >> 3) & 1;
    const int gate = gb >> 6;
    const int b = gb & 63;
    const float4* src = (const float4*)(xg_t + (size_t)b * NG_ + gate * H_ +
                                        wg * 16 + half * 8);
    float4* dst = (float4*)(g_lds + e);
    dst[0] = src[0];
    dst[1] = src[1];
  }
  __syncthreads();

  // 2) partial matmul: wave -> (gate, K-quarter); 8 K-iters x 4 WMMAs
  {
    const int lane = tid & 31;
    const int wave = tid >> 5;
    const int gate = wave & 3;
    const int kq = wave >> 2;
    const __bf16* Wrow = w_hh + (size_t)(gate * H_ + wg * 16) * H_;

    v8f acc[4] = {};
    for (int kk = 0; kk < 256; kk += 32) {
      const int k0 = kq * 256 + kk;
      const v16bf bmat = ldtile(Wrow, H_, k0, lane);
#pragma unroll
      for (int mi = 0; mi < 4; ++mi) {
        const v16bf amat = ldtile(h_in + (size_t)(mi * 16) * H_, H_, k0, lane);
        acc[mi] = __builtin_amdgcn_wmma_f32_16x16x32_bf16(
            false, amat, false, bmat, (short)0, acc[mi], false, false);
      }
    }
    const int col = lane & 15;
    const int mh = (lane >> 4) * 8;
#pragma unroll
    for (int mi = 0; mi < 4; ++mi)
#pragma unroll
      for (int r = 0; r < 8; ++r) {
        const int m = mi * 16 + mh + r;
        atomicAdd(&g_lds[(gate * 64 + m) * 16 + col], acc[mi][r]);  // ds_add_f32
      }
  }
  __syncthreads();

  // 3) pointwise gate math + state update (this WG's 16 columns only)
  for (int idx = tid; idx < B_ * 16; idx += 512) {
    const int b = idx >> 4;
    const int c = idx & 15;
    const float gi = g_lds[(0 * 64 + b) * 16 + c];
    const float gf = g_lds[(1 * 64 + b) * 16 + c];
    const float gg = g_lds[(2 * 64 + b) * 16 + c];
    const float go = g_lds[(3 * 64 + b) * 16 + c];
    const float i = sigmoidf_(gi);
    const float f = sigmoidf_(gf);
    const float g = tanhf(gg);
    const float o = sigmoidf_(go);
    const size_t hoff = (size_t)b * H_ + wg * 16 + c;
    const float cn = f * c_state[hoff] + i * g;
    const float hn = o * tanhf(cn);
    c_state[hoff] = cn;
    h_out[hoff] = (__bf16)hn;
    if (hseq_out) hseq_out[hoff] = (__bf16)hn;
    if (hout_f32) hout_f32[hoff] = hn;
  }
}

// ---------------------------------------------------------------------------
extern "C" void kernel_launch(void* const* d_in, const int* in_sizes, int n_in,
                              void* d_out, int out_size, void* d_ws,
                              size_t ws_size, hipStream_t stream) {
  const float* x     = (const float*)d_in[0];
  const float* w_ih0 = (const float*)d_in[1];
  const float* w_hh0 = (const float*)d_in[2];
  const float* b_ih0 = (const float*)d_in[3];
  const float* b_hh0 = (const float*)d_in[4];
  const float* w_ih1 = (const float*)d_in[5];
  const float* w_hh1 = (const float*)d_in[6];
  const float* b_ih1 = (const float*)d_in[7];
  const float* b_hh1 = (const float*)d_in[8];
  float* out = (float*)d_out;

  char* ws = (char*)d_ws;
  const size_t MB = 1024ull * 1024ull;
  __bf16* w_ih0_bf = (__bf16*)(ws + 0 * MB);        // 8 MB
  __bf16* w_hh0_bf = (__bf16*)(ws + 8 * MB);        // 8 MB
  __bf16* w_ih1_bf = (__bf16*)(ws + 16 * MB);       // 8 MB
  __bf16* w_hh1_bf = (__bf16*)(ws + 24 * MB);       // 8 MB
  __bf16* x_bf     = (__bf16*)(ws + 32 * MB);       // 64 MB  [T*B, D]
  __bf16* h1seq_bf = (__bf16*)(ws + 96 * MB);       // 64 MB  [T*B, H]
  float*  xg       = (float*)(ws + 160 * MB);       // 512 MB [T*B, NG]
  __bf16* hbuf0    = (__bf16*)(ws + 672 * MB);      // 128 KB
  __bf16* hbuf1    = (__bf16*)(ws + 672 * MB + 128 * 1024);
  float*  cbuf     = (float*)(ws + 672 * MB + 256 * 1024);  // 256 KB

  // fp32 -> bf16 conversions
  {
    const long nx = (long)T_ * B_ * D_;
    const long nw = (long)NG_ * D_;
    cvt_f32_bf16<<<(nx / 4 + 255) / 256, 256, 0, stream>>>(x, x_bf, nx / 4);
    cvt_f32_bf16<<<(nw / 4 + 255) / 256, 256, 0, stream>>>(w_ih0, w_ih0_bf, nw / 4);
    cvt_f32_bf16<<<(nw / 4 + 255) / 256, 256, 0, stream>>>(w_hh0, w_hh0_bf, nw / 4);
    cvt_f32_bf16<<<(nw / 4 + 255) / 256, 256, 0, stream>>>(w_ih1, w_ih1_bf, nw / 4);
    cvt_f32_bf16<<<(nw / 4 + 255) / 256, 256, 0, stream>>>(w_hh1, w_hh1_bf, nw / 4);
  }

  __bf16* hb[2] = {hbuf0, hbuf1};
  const dim3 ggrid(NG_ / 64, (T_ * B_) / (64 * 8));  // (64, 64)

  // ---------------- layer 0 ----------------
  gemm_xg<<<ggrid, 256, 0, stream>>>(x_bf, w_ih0_bf, b_ih0, b_hh0, xg);
  init_state<<<(B_ * H_) / 256, 256, 0, stream>>>(hbuf0, cbuf);
  for (int t = 0; t < T_; ++t) {
    lstm_step<<<H_ / 16, 512, 0, stream>>>(
        xg + (size_t)t * B_ * NG_, w_hh0_bf, hb[t & 1], hb[(t + 1) & 1], cbuf,
        h1seq_bf + (size_t)t * B_ * H_, (float*)nullptr);
  }

  // ---------------- layer 1 ----------------
  gemm_xg<<<ggrid, 256, 0, stream>>>(h1seq_bf, w_ih1_bf, b_ih1, b_hh1, xg);
  init_state<<<(B_ * H_) / 256, 256, 0, stream>>>(hbuf0, cbuf);
  for (int t = 0; t < T_; ++t) {
    lstm_step<<<H_ / 16, 512, 0, stream>>>(
        xg + (size_t)t * B_ * NG_, w_hh1_bf, hb[t & 1], hb[(t + 1) & 1], cbuf,
        (__bf16*)nullptr, out + (size_t)t * B_ * H_);
  }
}
